// SelfAttention2D_34299608826024
// MI455X (gfx1250) — matrix operand order repaired and checked by reading, not compiled, not run on hardware
//
#include <hip/hip_runtime.h>
#include <hip/hip_bf16.h>
#include <math.h>

typedef __attribute__((ext_vector_type(16))) __bf16 v16bf;
typedef __attribute__((ext_vector_type(8)))  __bf16 bf16x8;
typedef __attribute__((ext_vector_type(8)))  float  v8f;
typedef __attribute__((ext_vector_type(8)))  float  f32x8;

#define DEV static __device__ __forceinline__

// ---- fixed shapes from the reference ----
#define BATCH 4
#define NTOK  4096            // H*W
#define TOKENS (BATCH * NTOK) // 16384
#define CIN   256
#define RANK  32
#define NT_PER_B (NTOK / 16)  // 256

DEV v8f wmma_bf16(v16bf a, v16bf b, v8f c) {
  return __builtin_amdgcn_wmma_f32_16x16x32_bf16(false, a, false, b, (short)0, c,
                                                 false, false);
}

DEV v16bf cat8(bf16x8 lo, bf16x8 hi) {
  return __builtin_shufflevector(lo, hi, 0, 1, 2, 3, 4, 5, 6, 7,
                                 8, 9, 10, 11, 12, 13, 14, 15);
}

// A-matrix (16x32 bf16): lane=row(L&15), half=L>>4; elem i -> K=16*(i>>3)+8*half+(i&7).
// Per lane: two contiguous 16B chunks at K=8*half and K=16+8*half.
DEV v16bf load_A_bf16(const __bf16* base, int stride, int lane) {
  const int row = lane & 15, half = lane >> 4;
  const __bf16* p = base + (size_t)row * stride + (half << 3);
  bf16x8 lo = *(const bf16x8*)(p);
  bf16x8 hi = *(const bf16x8*)(p + 16);
  return cat8(lo, hi);
}

// A from f32 source (x): two contiguous 32B chunks, convert to bf16.
DEV v16bf load_A_f32(const float* base, int stride, int lane) {
  const int row = lane & 15, half = lane >> 4;
  const float* p = base + (size_t)row * stride + (half << 3);
  f32x8 lo = *(const f32x8*)(p);
  f32x8 hi = *(const f32x8*)(p + 16);
  v16bf a;
#pragma unroll
  for (int i = 0; i < 8; ++i) {
    a[i]     = (__bf16)lo[i];
    a[8 + i] = (__bf16)hi[i];
  }
  return a;
}

// B-matrix (32x16 bf16): lane=col(L&15), half=L>>4; elem i -> K=16*(i>>3)+8*half+(i&7).
// Source laid out K-contiguous per column: src[col][k] (i.e. Bmat[k][col]=src[col][k]).
// Per lane: two contiguous 16B chunks at K=8*half and K=16+8*half.
DEV v16bf load_B_kcontig(const __bf16* base, size_t stride, int lane) {
  const int col = lane & 15, half = lane >> 4;
  const __bf16* p = base + (size_t)col * stride + (half << 3);
  bf16x8 lo = *(const bf16x8*)(p);
  bf16x8 hi = *(const bf16x8*)(p + 16);
  return cat8(lo, hi);
}

// ---------------------------------------------------------------------------
// K0: one-time weight prep -> bf16, B-operand (K-contiguous) layouts.
//   wfT/wgT/whT : [RANK][CIN]  (wT[r][c] = w[c][r])
//   wvT         : [CIN][RANK]  (wvT[c][r] = wv[r][c])
__global__ void __launch_bounds__(256)
k_prep_w(const float* __restrict__ wf, const float* __restrict__ wg,
         const float* __restrict__ wh, const float* __restrict__ wv,
         __bf16* __restrict__ wfT, __bf16* __restrict__ wgT,
         __bf16* __restrict__ whT, __bf16* __restrict__ wvT) {
  const int i = blockIdx.x * blockDim.x + threadIdx.x;  // 0 .. 4*8192-1
  const int which = i >> 13;
  const int j = i & 8191;
  if (which < 3) {
    const float* w = (which == 0) ? wf : (which == 1) ? wg : wh;
    __bf16* wT = (which == 0) ? wfT : (which == 1) ? wgT : whT;
    const int r = j >> 8, c = j & 255;
    wT[j] = (__bf16)w[(size_t)c * RANK + r];
  } else {
    const int c = j >> 5, r = j & 31;
    wvT[j] = (__bf16)wv[(size_t)r * CIN + c];
  }
}

// ---------------------------------------------------------------------------
// K1: f,g,h = x @ {wf,wg,wh}.  One wave per 16-token tile.
__global__ void __launch_bounds__(32)
k_proj(const float* __restrict__ x, const __bf16* __restrict__ wfT,
       const __bf16* __restrict__ wgT, const __bf16* __restrict__ whT,
       __bf16* __restrict__ fb, __bf16* __restrict__ gb, float* __restrict__ hf) {
  const int lane = threadIdx.x & 31;
  const int t0 = blockIdx.x * 16;
  const float* xt = x + (size_t)t0 * CIN;

  v8f af0 = {}, af1 = {}, ag0 = {}, ag1 = {}, ah0 = {}, ah1 = {};
#pragma unroll 2
  for (int k0 = 0; k0 < CIN; k0 += 32) {
    v16bf a = load_A_f32(xt + k0, CIN, lane);
    af0 = wmma_bf16(a, load_B_kcontig(wfT + k0, CIN, lane), af0);
    af1 = wmma_bf16(a, load_B_kcontig(wfT + (size_t)16 * CIN + k0, CIN, lane), af1);
    ag0 = wmma_bf16(a, load_B_kcontig(wgT + k0, CIN, lane), ag0);
    ag1 = wmma_bf16(a, load_B_kcontig(wgT + (size_t)16 * CIN + k0, CIN, lane), ag1);
    ah0 = wmma_bf16(a, load_B_kcontig(whT + k0, CIN, lane), ah0);
    ah1 = wmma_bf16(a, load_B_kcontig(whT + (size_t)16 * CIN + k0, CIN, lane), ah1);
  }
  const int col = lane & 15, half = lane >> 4;
#pragma unroll
  for (int v = 0; v < 8; ++v) {
    const size_t r = (size_t)(t0 + v + (half << 3)) * RANK;
    fb[r + col]      = (__bf16)af0[v];
    fb[r + 16 + col] = (__bf16)af1[v];
    gb[r + col]      = (__bf16)ag0[v];
    gb[r + 16 + col] = (__bf16)ag1[v];
    hf[r + col]      = ah0[v];
    hf[r + 16 + col] = ah1[v];
  }
}

// ---------------------------------------------------------------------------
// K2: column softmax stats over the query axis n (softmax(axis=1)):
//     M[m]=max_n s[n,m], Z[m]=sum_n exp(s[n,m]-M[m]),  s = f @ g^T.
__global__ void __launch_bounds__(32)
k_stats(const __bf16* __restrict__ fb, const __bf16* __restrict__ gb,
        float* __restrict__ Mv, float* __restrict__ Zv) {
  const int lane = threadIdx.x & 31;
  const int b  = blockIdx.x / NT_PER_B;
  const int m0 = (blockIdx.x % NT_PER_B) * 16;
  const size_t base = (size_t)b * NTOK;

  // g rows are K(=r)-contiguous -> direct B operand for f @ g^T
  const v16bf bg = load_B_kcontig(gb + (base + m0) * RANK, RANK, lane);

  float mrun = -INFINITY, zrun = 0.0f;
  for (int n0 = 0; n0 < NTOK; n0 += 16) {
    __builtin_prefetch(fb + (base + n0 + 64) * RANK, 0, 0);
    v16bf a = load_A_bf16(fb + (base + n0) * RANK, RANK, lane);
    v8f s = {};
    s = wmma_bf16(a, bg, s);
    float tmax = s[0];
#pragma unroll
    for (int v = 1; v < 8; ++v) tmax = fmaxf(tmax, s[v]);
    const float nm = fmaxf(mrun, tmax);
    float z = zrun * __expf(mrun - nm);
#pragma unroll
    for (int v = 0; v < 8; ++v) z += __expf(s[v] - nm);
    mrun = nm;
    zrun = z;
  }
  const float mo = __shfl_xor(mrun, 16, 32);
  const float zo = __shfl_xor(zrun, 16, 32);
  const float M = fmaxf(mrun, mo);
  const float Z = zrun * __expf(mrun - M) + zo * __expf(mo - M);
  if (lane < 16) {
    Mv[base + m0 + lane] = M;
    Zv[base + m0 + lane] = Z;
  }
}

// ---------------------------------------------------------------------------
// K3: hnT[r][t] = h[t][r] / Z[t]  -> bf16, transposed so pass-2 B is K-contiguous.
__global__ void __launch_bounds__(256)
k_scale_h(const float* __restrict__ hf, const float* __restrict__ Zv,
          __bf16* __restrict__ hnT, int total) {
  const int j = blockIdx.x * blockDim.x + threadIdx.x;  // coalesced over tokens
  if (j < total) {
    const int r = j >> 14;          // j / TOKENS
    const int t = j & (TOKENS - 1); // j % TOKENS
    hnT[j] = (__bf16)(hf[(size_t)t * RANK + r] / Zv[t]);
  }
}

// ---------------------------------------------------------------------------
// K4: y[n,:] = sum_m exp(s[n,m]-M[m]) * hn[m,:]   (scores recomputed, never stored)
__global__ void __launch_bounds__(32)
k_attn(const __bf16* __restrict__ fb, const __bf16* __restrict__ gb,
       const __bf16* __restrict__ hnT, const float* __restrict__ Mv,
       __bf16* __restrict__ yb) {
  __shared__ __align__(16) __bf16 Pls[16 * 32];
  const int lane = threadIdx.x & 31;
  const int b  = blockIdx.x / NT_PER_B;
  const int n0 = (blockIdx.x % NT_PER_B) * 16;
  const size_t base = (size_t)b * NTOK;
  const int col = lane & 15, half = lane >> 4;

  const v16bf af = load_A_bf16(fb + (base + n0) * RANK, RANK, lane);
  v8f y0 = {}, y1 = {};

  for (int m0 = 0; m0 < NTOK; m0 += 32) {
    __builtin_prefetch(gb + (base + m0 + 64) * RANK, 0, 0);
    __builtin_prefetch(hnT + base + m0 + 64, 0, 0);
#pragma unroll
    for (int sub = 0; sub < 2; ++sub) {
      const int mm = m0 + sub * 16;
      v16bf bgt = load_B_kcontig(gb + (base + mm) * RANK, RANK, lane);
      v8f s = {};
      s = wmma_bf16(af, bgt, s);                 // s[n-tile, m-tile]
      const float Mc = Mv[base + mm + col];
#pragma unroll
      for (int v = 0; v < 8; ++v)                // C-layout -> row-major P in LDS
        Pls[(v + (half << 3)) * 32 + sub * 16 + col] = (__bf16)__expf(s[v] - Mc);
    }
    __syncthreads();
    v16bf ap  = load_A_bf16(Pls, 32, lane);      // P 16x32 (ds_load_b128 x2)
    // hnT: Bmat[k][r] = hnT[r][base+m0+k] -> K-contiguous rows of length TOKENS
    v16bf bh0 = load_B_kcontig(hnT + base + m0, TOKENS, lane);
    v16bf bh1 = load_B_kcontig(hnT + (size_t)16 * TOKENS + base + m0, TOKENS, lane);
    y0 = wmma_bf16(ap, bh0, y0);
    y1 = wmma_bf16(ap, bh1, y1);
    __syncthreads();
  }
#pragma unroll
  for (int v = 0; v < 8; ++v) {
    const size_t r = (base + n0 + v + (half << 3)) * RANK;
    yb[r + col]      = (__bf16)y0[v];
    yb[r + 16 + col] = (__bf16)y1[v];
  }
}

// ---------------------------------------------------------------------------
// K5: out = gamma * (y @ w_v) + x
__global__ void __launch_bounds__(32)
k_out(const __bf16* __restrict__ yb, const __bf16* __restrict__ wvT,
      const float* __restrict__ x, const float* __restrict__ gamma,
      float* __restrict__ out) {
  const int lane = threadIdx.x & 31;
  const int t0 = blockIdx.x * 16;
  const int col = lane & 15, half = lane >> 4;
  const float gm = gamma[0];

  const v16bf ay = load_A_bf16(yb + (size_t)t0 * RANK, RANK, lane);
#pragma unroll 4
  for (int ct = 0; ct < CIN / 16; ++ct) {
    v16bf bw = load_B_kcontig(wvT + (size_t)(ct * 16) * RANK, RANK, lane);
    v8f c = {};
    c = wmma_bf16(ay, bw, c);
#pragma unroll
    for (int v = 0; v < 8; ++v) {
      const size_t idx = (size_t)(t0 + v + (half << 3)) * CIN + ct * 16 + col;
      out[idx] = gm * c[v] + x[idx];
    }
  }
}

// ---------------------------------------------------------------------------
extern "C" void kernel_launch(void* const* d_in, const int* in_sizes, int n_in,
                              void* d_out, int out_size, void* d_ws, size_t ws_size,
                              hipStream_t stream) {
  const float* x     = (const float*)d_in[0];
  const float* w_f   = (const float*)d_in[1];
  const float* w_g   = (const float*)d_in[2];
  const float* w_h   = (const float*)d_in[3];
  const float* w_v   = (const float*)d_in[4];
  const float* gamma = (const float*)d_in[5];
  float* out = (float*)d_out;
  (void)in_sizes; (void)n_in; (void)out_size; (void)ws_size;

  const size_t MB = 1u << 20;
  char* ws = (char*)d_ws;
  __bf16* fb  = (__bf16*)(ws);                          // 1 MB  [TOKENS][32] bf16
  __bf16* gb  = (__bf16*)(ws + 1 * MB);                 // 1 MB
  float*  hf  = (float*)(ws + 2 * MB);                  // 2 MB  [TOKENS][32] f32
  float*  Mv  = (float*)(ws + 4 * MB);                  // 64 KB
  float*  Zv  = (float*)(ws + 4 * MB + (64u << 10));    // 64 KB
  __bf16* hnT = (__bf16*)(ws + 4 * MB + (128u << 10));  // 1 MB  [32][TOKENS] bf16
  __bf16* yb  = (__bf16*)(ws + 5 * MB + (128u << 10));  // 1 MB  [TOKENS][32] bf16
  char* wbase = ws + 6 * MB + (128u << 10);
  __bf16* wfT = (__bf16*)(wbase);                       // 16 KB [32][256]
  __bf16* wgT = (__bf16*)(wbase + (16u << 10));         // 16 KB
  __bf16* whT = (__bf16*)(wbase + (32u << 10));         // 16 KB
  __bf16* wvT = (__bf16*)(wbase + (48u << 10));         // 16 KB [256][32]

  const int tileBlocks = TOKENS / 16;                   // 1024

  k_prep_w<<<(4 * 8192) / 256, 256, 0, stream>>>(w_f, w_g, w_h, w_v,
                                                 wfT, wgT, whT, wvT);
  k_proj<<<tileBlocks, 32, 0, stream>>>(x, wfT, wgT, whT, fb, gb, hf);
  k_stats<<<tileBlocks, 32, 0, stream>>>(fb, gb, Mv, Zv);
  k_scale_h<<<(TOKENS * RANK + 255) / 256, 256, 0, stream>>>(hf, Zv, hnT,
                                                             TOKENS * RANK);
  k_attn<<<tileBlocks, 32, 0, stream>>>(fb, gb, hnT, Mv, yb);
  k_out<<<tileBlocks, 32, 0, stream>>>(yb, wvT, x, gamma, out);
}
